// MemPIDModel_31525059952782
// MI455X (gfx1250) — compile-verified, matrix operand order checked by reading, not run on hardware
//
#include <hip/hip_runtime.h>
#include <math.h>

#define DIMc   512
#define LAYERSc 6
#define Kc     15
#define VOCABc 32000
#define Bc     2
#define Lc     1024
#define HIDc   1024
#define RANKc  64
#define MTOK   (Bc * Lc)        // 2048
#define MIXW   0.1f
#define PADh   8                // LDS row padding (halves) -> 16B/row rotation

typedef __attribute__((ext_vector_type(16))) _Float16 v16h;
typedef __attribute__((ext_vector_type(8)))  _Float16 v8h;
typedef __attribute__((ext_vector_type(4)))  _Float16 v4h;
typedef __attribute__((ext_vector_type(8)))  float    v8f;
typedef int v4i __attribute__((vector_size(4 * sizeof(int))));

#if defined(__has_builtin)
#  if __has_builtin(__builtin_amdgcn_global_load_async_to_lds_b128)
#    define HAVE_ASYNC 1
#  endif
#endif
#ifndef HAVE_ASYNC
#  define HAVE_ASYNC 0
#endif

// ---------------------------------------------------------------------------
// helpers
// ---------------------------------------------------------------------------
__device__ __forceinline__ float silu_f(float x)    { return x / (1.0f + __expf(-x)); }
__device__ __forceinline__ float sigmoid_f(float x) { return 1.0f / (1.0f + __expf(-x)); }

__device__ __forceinline__ v8f wmma16x16x32(v16h a, v16h b, v8f c) {
    return __builtin_amdgcn_wmma_f32_16x16x32_f16(false, a, false, b, (short)0, c,
                                                  false, false);
}

__device__ __forceinline__ v16h combine8(v8h lo, v8h hi) {
    return __builtin_shufflevector(lo, hi, 0, 1, 2, 3, 4, 5, 6, 7,
                                           8, 9, 10, 11, 12, 13, 14, 15);
}

// Per-lane operand pointer for a 16x32 f16 WMMA tile from a row-major [out,in]
// matrix: lane&15 = row/col, lane>>4 selects the K-half.
__device__ __forceinline__ const _Float16* tile_ptr(const _Float16* base, int ld,
                                                    int rc0, int lane) {
    return base + (size_t)(rc0 + (lane & 15)) * (size_t)ld + (lane >> 4) * 8;
}
__device__ __forceinline__ v16h tile_ld(const _Float16* p) {
    return combine8(*(const v8h*)p, *(const v8h*)(p + 16));
}

// Cooperative load of a 32-row x K A-strip into padded LDS.
// Uses gfx1250 async global->LDS when available; plain ds stores otherwise.
template <int K>
__device__ __forceinline__ void strip_load(_Float16* sA,
                                           const _Float16* __restrict__ X,
                                           int row0) {
    const int tid = threadIdx.x;                 // 256 threads
    constexpr int CH   = K / 8;                  // 16-byte chunks per row
    constexpr int ITER = (32 * CH) / 256;
#pragma unroll
    for (int j = 0; j < ITER; ++j) {
        int c   = tid + j * 256;
        int row = c / CH;
        int kc  = (c % CH) * 8;
        const _Float16* src = X + (size_t)(row0 + row) * K + kc;
        _Float16*       dst = sA + row * (K + PADh) + kc;
#if HAVE_ASYNC
        __builtin_amdgcn_global_load_async_to_lds_b128((v4i*)src, (v4i*)dst, 0, 0);
#else
        *(v8h*)dst = *(const v8h*)src;
#endif
    }
#if HAVE_ASYNC
#  if __has_builtin(__builtin_amdgcn_s_wait_asynccnt)
    __builtin_amdgcn_s_wait_asynccnt(0);
#  else
    asm volatile("s_wait_asynccnt 0x0" ::: "memory");
#  endif
#endif
    __syncthreads();
}

// ---------------------------------------------------------------------------
// GEMM kernels:  Y[M,N] = X[M,K] @ W[N,K]^T   (f16 operands, f32 accum)
// block = 256 threads = 8 waves; each wave computes a 32 x (16*NT) tile.
// ---------------------------------------------------------------------------

// H = silu(X@W1^T) * (X@W3^T) -> f16   (K=512, N=1024; A staged in LDS)
__global__ void __launch_bounds__(256)
gemm_dual_silu(const _Float16* __restrict__ X, const _Float16* __restrict__ W1,
               const _Float16* __restrict__ W3, _Float16* __restrict__ H, int N) {
    __shared__ _Float16 sA[32 * (DIMc + PADh)];
    const int row0 = blockIdx.x * 32;
    strip_load<DIMc>(sA, X, row0);
    const int lane = threadIdx.x & 31;
    const int wave = threadIdx.x >> 5;
    const int col0 = (blockIdx.y * 8 + wave) * 32;
    const int m = lane & 15, half = lane >> 4;
    const _Float16* sa0 = sA + m * (DIMc + PADh) + half * 8;
    const _Float16* sa1 = sa0 + 16 * (DIMc + PADh);
    const _Float16* p10 = tile_ptr(W1, DIMc, col0,      lane);
    const _Float16* p11 = tile_ptr(W1, DIMc, col0 + 16, lane);
    const _Float16* p30 = tile_ptr(W3, DIMc, col0,      lane);
    const _Float16* p31 = tile_ptr(W3, DIMc, col0 + 16, lane);
    v8f c1[2][2] = {}; v8f c3[2][2] = {};
    for (int k0 = 0; k0 < DIMc; k0 += 32) {
        v16h a0  = combine8(*(const v8h*)sa0, *(const v8h*)(sa0 + 16));
        v16h a1  = combine8(*(const v8h*)sa1, *(const v8h*)(sa1 + 16));
        v16h b10 = tile_ld(p10);
        v16h b11 = tile_ld(p11);
        v16h b30 = tile_ld(p30);
        v16h b31 = tile_ld(p31);
        if (k0 + 64 < DIMc) {                    // gfx1250 global_prefetch_b8
            __builtin_prefetch(p10 + 64, 0, 3);
            __builtin_prefetch(p11 + 64, 0, 3);
            __builtin_prefetch(p30 + 64, 0, 3);
            __builtin_prefetch(p31 + 64, 0, 3);
        }
        c1[0][0] = wmma16x16x32(a0, b10, c1[0][0]);
        c1[1][0] = wmma16x16x32(a1, b10, c1[1][0]);
        c1[0][1] = wmma16x16x32(a0, b11, c1[0][1]);
        c1[1][1] = wmma16x16x32(a1, b11, c1[1][1]);
        c3[0][0] = wmma16x16x32(a0, b30, c3[0][0]);
        c3[1][0] = wmma16x16x32(a1, b30, c3[1][0]);
        c3[0][1] = wmma16x16x32(a0, b31, c3[0][1]);
        c3[1][1] = wmma16x16x32(a1, b31, c3[1][1]);
        sa0 += 32; sa1 += 32; p10 += 32; p11 += 32; p30 += 32; p31 += 32;
    }
    const int n = lane & 15;
#pragma unroll
    for (int mt = 0; mt < 2; ++mt)
#pragma unroll
        for (int t = 0; t < 2; ++t)
#pragma unroll
            for (int r = 0; r < 8; ++r) {
                float v1 = c1[mt][t][r], v3 = c3[mt][t][r];
                size_t o = (size_t)(row0 + mt * 16 + half * 8 + r) * (size_t)N +
                           col0 + 16 * t + n;
                H[o] = (_Float16)(silu_f(v1) * v3);
            }
}

// OUT(f32) = X@W^T + RES(f32)          (K=1024, N=512; MT=2, direct A)
__global__ void __launch_bounds__(256)
gemm_res(const _Float16* __restrict__ X, const _Float16* __restrict__ W,
         const float* __restrict__ RES, float* __restrict__ OUT, int N) {
    const int lane = threadIdx.x & 31;
    const int wave = threadIdx.x >> 5;
    const int row0 = blockIdx.x * 32;
    const int col0 = (blockIdx.y * 8 + wave) * 32;
    const _Float16* pa0 = tile_ptr(X, HIDc, row0,      lane);
    const _Float16* pa1 = tile_ptr(X, HIDc, row0 + 16, lane);
    const _Float16* p0  = tile_ptr(W, HIDc, col0,      lane);
    const _Float16* p1  = tile_ptr(W, HIDc, col0 + 16, lane);
    v8f c[2][2] = {};
    for (int k0 = 0; k0 < HIDc; k0 += 32) {
        v16h a0 = tile_ld(pa0);
        v16h a1 = tile_ld(pa1);
        v16h b0 = tile_ld(p0);
        v16h b1 = tile_ld(p1);
        if (k0 + 64 < HIDc) {
            __builtin_prefetch(p0 + 64, 0, 3);
            __builtin_prefetch(p1 + 64, 0, 3);
        }
        c[0][0] = wmma16x16x32(a0, b0, c[0][0]);
        c[1][0] = wmma16x16x32(a1, b0, c[1][0]);
        c[0][1] = wmma16x16x32(a0, b1, c[0][1]);
        c[1][1] = wmma16x16x32(a1, b1, c[1][1]);
        pa0 += 32; pa1 += 32; p0 += 32; p1 += 32;
    }
    const int n = lane & 15, half = lane >> 4;
#pragma unroll
    for (int mt = 0; mt < 2; ++mt)
#pragma unroll
        for (int t = 0; t < 2; ++t)
#pragma unroll
            for (int r = 0; r < 8; ++r) {
                size_t o = (size_t)(row0 + mt * 16 + half * 8 + r) * (size_t)N +
                           col0 + 16 * t + n;
                OUT[o] = c[mt][t][r] + RES[o];
            }
}

// OUT(f32) = X@W^T   (K=512, N=32000; A in LDS, MT=2/NT=4, clamped tail)
__global__ void __launch_bounds__(256)
gemm_logits(const _Float16* __restrict__ X, const _Float16* __restrict__ W,
            float* __restrict__ OUT, int N) {
    __shared__ _Float16 sA[32 * (DIMc + PADh)];
    const int row0 = blockIdx.x * 32;
    strip_load<DIMc>(sA, X, row0);
    const int lane = threadIdx.x & 31;
    const int wave = threadIdx.x >> 5;
    int col0 = (blockIdx.y * 8 + wave) * 64;
    if (col0 + 64 > N) col0 = N - 64;            // wave-uniform clamp (benign dup)
    const int m = lane & 15, half = lane >> 4;
    const _Float16* sa0 = sA + m * (DIMc + PADh) + half * 8;
    const _Float16* sa1 = sa0 + 16 * (DIMc + PADh);
    const _Float16* p0 = tile_ptr(W, DIMc, col0,      lane);
    const _Float16* p1 = tile_ptr(W, DIMc, col0 + 16, lane);
    const _Float16* p2 = tile_ptr(W, DIMc, col0 + 32, lane);
    const _Float16* p3 = tile_ptr(W, DIMc, col0 + 48, lane);
    v8f c[2][4] = {};
    for (int k0 = 0; k0 < DIMc; k0 += 32) {
        v16h a0 = combine8(*(const v8h*)sa0, *(const v8h*)(sa0 + 16));
        v16h a1 = combine8(*(const v8h*)sa1, *(const v8h*)(sa1 + 16));
        v16h b0 = tile_ld(p0);
        v16h b1 = tile_ld(p1);
        v16h b2 = tile_ld(p2);
        v16h b3 = tile_ld(p3);
        if (k0 + 64 < DIMc) {
            __builtin_prefetch(p0 + 64, 0, 3);
            __builtin_prefetch(p1 + 64, 0, 3);
            __builtin_prefetch(p2 + 64, 0, 3);
            __builtin_prefetch(p3 + 64, 0, 3);
        }
        c[0][0] = wmma16x16x32(a0, b0, c[0][0]);
        c[1][0] = wmma16x16x32(a1, b0, c[1][0]);
        c[0][1] = wmma16x16x32(a0, b1, c[0][1]);
        c[1][1] = wmma16x16x32(a1, b1, c[1][1]);
        c[0][2] = wmma16x16x32(a0, b2, c[0][2]);
        c[1][2] = wmma16x16x32(a1, b2, c[1][2]);
        c[0][3] = wmma16x16x32(a0, b3, c[0][3]);
        c[1][3] = wmma16x16x32(a1, b3, c[1][3]);
        sa0 += 32; sa1 += 32; p0 += 32; p1 += 32; p2 += 32; p3 += 32;
    }
    const int n = lane & 15;
#pragma unroll
    for (int mt = 0; mt < 2; ++mt)
#pragma unroll
        for (int t = 0; t < 4; ++t)
#pragma unroll
            for (int r = 0; r < 8; ++r)
                OUT[(size_t)(row0 + mt * 16 + half * 8 + r) * (size_t)N +
                    col0 + 16 * t + n] = c[mt][t][r];
}

// ---------------------------------------------------------------------------
// conversion + elementwise / reduction kernels
// ---------------------------------------------------------------------------
__global__ void cvt_f32_f16(const float* __restrict__ in,
                            _Float16* __restrict__ out, int n) {
    int i = (blockIdx.x * blockDim.x + threadIdx.x) * 4;
    if (i >= n) return;
    float4 v = *(const float4*)(in + i);
    v4h o;
    o[0] = (_Float16)v.x; o[1] = (_Float16)v.y;
    o[2] = (_Float16)v.z; o[3] = (_Float16)v.w;
    *(v4h*)(out + i) = o;
}

__global__ void embed_kernel(const int* __restrict__ idx,
                             const float* __restrict__ emb,
                             const float* __restrict__ pos,
                             float* __restrict__ out) {
    int id = blockIdx.x * blockDim.x + threadIdx.x;      // MTOK*DIM threads
    int d = id & (DIMc - 1);
    int tok = id >> 9;
    int t = tok & (Lc - 1);
    out[id] = emb[(size_t)idx[tok] * DIMc + d] + pos[(size_t)t * DIMc + d];
}

__global__ void rmsnorm_kernel(const float* __restrict__ x,
                               const float* __restrict__ w,
                               float* __restrict__ out, int ntok) {
    int tok  = blockIdx.x * 8 + (threadIdx.x >> 5);
    int lane = threadIdx.x & 31;
    if (tok >= ntok) return;
    const float* xp = x + (size_t)tok * DIMc;
    float v[16]; float s = 0.f;
#pragma unroll
    for (int j = 0; j < 16; ++j) { v[j] = xp[lane + 32 * j]; s += v[j] * v[j]; }
#pragma unroll
    for (int off = 16; off >= 1; off >>= 1) s += __shfl_xor(s, off, 32);
    float scale = rsqrtf(s * (1.0f / DIMc) + 1e-6f);
    float* op = out + (size_t)tok * DIMc;
#pragma unroll
    for (int j = 0; j < 16; ++j) { int d = lane + 32 * j; op[d] = v[j] * scale * w[d]; }
}

__global__ void rmsnorm_f16_kernel(const float* __restrict__ x,
                                   const float* __restrict__ w,
                                   _Float16* __restrict__ out, int ntok) {
    int tok  = blockIdx.x * 8 + (threadIdx.x >> 5);
    int lane = threadIdx.x & 31;
    if (tok >= ntok) return;
    const float* xp = x + (size_t)tok * DIMc;
    float v[16]; float s = 0.f;
#pragma unroll
    for (int j = 0; j < 16; ++j) { v[j] = xp[lane + 32 * j]; s += v[j] * v[j]; }
#pragma unroll
    for (int off = 16; off >= 1; off >>= 1) s += __shfl_xor(s, off, 32);
    float scale = rsqrtf(s * (1.0f / DIMc) + 1e-6f);
    _Float16* op = out + (size_t)tok * DIMc;
#pragma unroll
    for (int j = 0; j < 16; ++j) {
        int d = lane + 32 * j;
        op[d] = (_Float16)(v[j] * scale * w[d]);
    }
}

__global__ void memgate_kernel(const float* __restrict__ mp,
                               const float* __restrict__ gw,
                               const float* __restrict__ gb,
                               float* __restrict__ mem) {
    int id = blockIdx.x * blockDim.x + threadIdx.x;      // B*DIM
    int b = id >> 9, d = id & (DIMc - 1);
    const float* wp = gw + (size_t)d * DIMc;
    const float* pp = mp + (size_t)b * DIMc;
    float acc = gb[d];
    for (int e = 0; e < DIMc; ++e) acc += pp[e] * wp[e];
    mem[id] = sigmoid_f(acc);
}

__global__ void shift_add_kernel(const float* __restrict__ x,
                                 const float* __restrict__ ts,
                                 const float* __restrict__ mem,
                                 float* __restrict__ out) {
    int id = blockIdx.x * blockDim.x + threadIdx.x;
    int d = id & (DIMc - 1);
    int tok = id >> 9;
    int t = tok & (Lc - 1);
    int b = tok >> 10;
    float cur = x[id];
    float v = (t == 0) ? cur : ts[d] * x[id - DIMc] + (1.0f - ts[d]) * cur;
    out[id] = v + mem[b * DIMc + d];
}

__global__ void pid_kernel(const float* __restrict__ cur,
                           const float* __restrict__ prev,
                           const float* __restrict__ integ,
                           const float* __restrict__ kp,
                           const float* __restrict__ ki,
                           const float* __restrict__ kd,
                           const float* __restrict__ gn,
                           float inv_i, float* __restrict__ out) {
    int tok  = blockIdx.x * 8 + (threadIdx.x >> 5);
    int lane = threadIdx.x & 31;
    const float* cp = cur   + (size_t)tok * DIMc;
    const float* pp = prev  + (size_t)tok * DIMc;
    const float* ip = integ + (size_t)tok * DIMc;
    float sv[16]; float s = 0.f;
#pragma unroll
    for (int j = 0; j < 16; ++j) {
        int d = lane + 32 * j;
        float c = cp[d];
        float v = kp[d] * c + ki[d] * ip[d] * inv_i + kd[d] * (c - pp[d]);
        float u = silu_f(v);
        sv[j] = u; s += u * u;
    }
#pragma unroll
    for (int off = 16; off >= 1; off >>= 1) s += __shfl_xor(s, off, 32);
    float scale = rsqrtf(s * (1.0f / DIMc) + 1e-6f);
    float* op = out + (size_t)tok * DIMc;
#pragma unroll
    for (int j = 0; j < 16; ++j) { int d = lane + 32 * j; op[d] = sv[j] * scale * gn[d]; }
}

__global__ void conv_kernel(const float* __restrict__ h,
                            const float* __restrict__ cw,
                            const float* __restrict__ cb,
                            int dil, _Float16* __restrict__ out) {
    int id = blockIdx.x * blockDim.x + threadIdx.x;
    int d = id & (DIMc - 1);
    int tok = id >> 9;
    int t = tok & (Lc - 1);
    int b = tok >> 10;
    const float* wp = cw + (size_t)d * Kc;
    float acc = cb[d];
#pragma unroll
    for (int j = 0; j < Kc; ++j) {
        int tt = t - (Kc - 1 - j) * dil;
        if (tt >= 0)
            acc += wp[j] * h[((size_t)(b * Lc + tt)) * DIMc + d];
    }
    out[id] = (_Float16)silu_f(acc);
}

__global__ void update_kernel(const float* __restrict__ cur,
                              float* __restrict__ prev,
                              float* __restrict__ integ) {
    int id = blockIdx.x * blockDim.x + threadIdx.x;
    float c = cur[id];
    prev[id] = c;
    integ[id] += c;
}

__global__ void mix_kernel(float* __restrict__ x) {
    int id = blockIdx.x * blockDim.x + threadIdx.x;      // B*DIM
    int b = id >> 9, d = id & (DIMc - 1);
    float run = 0.f;
    for (int t = 0; t < Lc; ++t) {
        size_t o = ((size_t)(b * Lc + t)) * DIMc + d;
        float v = x[o];
        run += v;
        x[o] = v + MIXW * run / (float)(t + 1);
    }
}

__global__ void sgate_kernel(const float* __restrict__ nb,
                             const float* __restrict__ ob,
                             const float* __restrict__ nw,
                             const float* __restrict__ dw,
                             const float* __restrict__ db,
                             const float* __restrict__ uw,
                             const float* __restrict__ ub,
                             float* __restrict__ out) {
    __shared__ float rn[DIMc];
    __shared__ float hh[RANKc];
    __shared__ float red[16];
    int tok = blockIdx.x;
    int d = threadIdx.x;                                 // blockDim = 512
    const float* np = nb + (size_t)tok * DIMc;
    float v = np[d];
    float s = v * v;
#pragma unroll
    for (int off = 16; off >= 1; off >>= 1) s += __shfl_xor(s, off, 32);
    if ((d & 31) == 0) red[d >> 5] = s;
    __syncthreads();
    if (d == 0) {
        float t = 0.f;
        for (int i = 0; i < 16; ++i) t += red[i];
        red[0] = rsqrtf(t * (1.0f / DIMc) + 1e-6f);
    }
    __syncthreads();
    rn[d] = v * red[0] * nw[d];
    __syncthreads();
    if (d < RANKc) {
        const float* wp = dw + (size_t)d * DIMc;
        float acc = db[d];
        for (int j = 0; j < DIMc; ++j) acc += rn[j] * wp[j];
        hh[d] = silu_f(acc);
    }
    __syncthreads();
    const float* up = uw + (size_t)d * RANKc;
    float acc = ub[d];
#pragma unroll
    for (int r = 0; r < RANKc; ++r) acc += hh[r] * up[r];
    float g = sigmoid_f(acc);
    size_t o = (size_t)tok * DIMc + d;
    out[o] = g * v + (1.0f - g) * ob[o];
}

// ---------------------------------------------------------------------------
// host orchestration
// ---------------------------------------------------------------------------
struct StackP {
    const float *nw, *cw, *cb, *kp, *ki, *kd, *gn;
    const _Float16 *w1, *w2, *w3;
};

static void run_stack(const float* x_in, const int* dils, const StackP& P,
                      float* cur, float* prev, float* integ, float* curin,
                      float* t1, _Float16* t2h, _Float16* hidh, hipStream_t s) {
    const size_t bytes = (size_t)MTOK * DIMc * sizeof(float);
    (void)hipMemcpyAsync(cur,   x_in, bytes, hipMemcpyDeviceToDevice, s);
    (void)hipMemcpyAsync(prev,  x_in, bytes, hipMemcpyDeviceToDevice, s);
    (void)hipMemcpyAsync(integ, x_in, bytes, hipMemcpyDeviceToDevice, s);
    const int nEl = MTOK * DIMc;
    for (int i = 0; i < LAYERSc; ++i) {
        if (i == 0) {
            (void)hipMemcpyAsync(curin, cur, bytes, hipMemcpyDeviceToDevice, s);
        } else {
            pid_kernel<<<MTOK / 8, 256, 0, s>>>(
                cur, prev, integ,
                P.kp + (size_t)(i - 1) * DIMc, P.ki + (size_t)(i - 1) * DIMc,
                P.kd + (size_t)(i - 1) * DIMc, P.gn + (size_t)(i - 1) * DIMc,
                1.0f / (float)i, curin);
        }
        rmsnorm_kernel<<<MTOK / 8, 256, 0, s>>>(curin, P.nw + (size_t)i * DIMc, t1, MTOK);
        conv_kernel<<<nEl / 256, 256, 0, s>>>(t1, P.cw + (size_t)i * DIMc * Kc,
                                              P.cb + (size_t)i * DIMc, dils[i], t2h);
        dim3 g1(MTOK / 32, HIDc / 256);                  // (64, 4)
        gemm_dual_silu<<<g1, 256, 0, s>>>(t2h, P.w1 + (size_t)i * HIDc * DIMc,
                                          P.w3 + (size_t)i * HIDc * DIMc,
                                          hidh, HIDc);
        dim3 g2(MTOK / 32, DIMc / 256);                  // (64, 2)
        gemm_res<<<g2, 256, 0, s>>>(hidh, P.w2 + (size_t)i * DIMc * HIDc,
                                    curin, cur, DIMc);
        update_kernel<<<nEl / 256, 256, 0, s>>>(cur, prev, integ);
    }
}

extern "C" void kernel_launch(void* const* d_in, const int* in_sizes, int n_in,
                              void* d_out, int out_size, void* d_ws, size_t ws_size,
                              hipStream_t stream) {
    const int*   idx         = (const int*)  d_in[0];
    const float* emb         = (const float*)d_in[1];
    const float* pos         = (const float*)d_in[2];
    const float* emb_norm_w  = (const float*)d_in[3];
    const float* token_shift = (const float*)d_in[4];
    const float* mem_gate_w  = (const float*)d_in[5];
    const float* mem_gate_b  = (const float*)d_in[6];
    const float* memory_p    = (const float*)d_in[7];
    const float* sg_norm     = (const float*)d_in[8];
    const float* sg_down_w   = (const float*)d_in[9];
    const float* sg_down_b   = (const float*)d_in[10];
    const float* sg_up_w     = (const float*)d_in[11];
    const float* sg_up_b     = (const float*)d_in[12];
    const float* final_norm_w= (const float*)d_in[13];
    const float* up_f32[10]; const float* dn_f32[10];
    for (int i = 0; i < 10; ++i) { up_f32[i] = (const float*)d_in[14 + i];
                                   dn_f32[i] = (const float*)d_in[24 + i]; }

    const size_t NE = (size_t)MTOK * DIMc;               // 1,048,576
    float* ws      = (float*)d_ws;
    float* initial = ws;
    float* cur     = ws + 1 * NE;
    float* prev    = ws + 2 * NE;
    float* integ   = ws + 3 * NE;
    float* curin   = ws + 4 * NE;
    float* t1      = ws + 5 * NE;
    float* g1b     = ws + 6 * NE;
    float* g2b     = ws + 7 * NE;
    float* ob      = ws + 8 * NE;
    float* mem     = ws + 9 * NE;
    _Float16* hb   = (_Float16*)(ws + 10 * NE);
    const size_t WSZ = 3 * NE;
    _Float16* t2h  = hb;
    _Float16* hidh = hb + 1 * NE;
    _Float16* t1h  = hb + 3 * NE;
    _Float16* w1u  = hb + 4 * NE;
    _Float16* w2u  = w1u + WSZ;
    _Float16* w3u  = w2u + WSZ;
    _Float16* w1d  = w3u + WSZ;
    _Float16* w2d  = w1d + WSZ;
    _Float16* w3d  = w2d + WSZ;
    _Float16* embh = w3d + WSZ;

    const int nEl = MTOK * DIMc;
    static const int UPD[LAYERSc] = {1, 2, 4, 8, 16, 32};
    static const int DND[LAYERSc] = {32, 16, 8, 4, 2, 1};

    const int WN = LAYERSc * HIDc * DIMc;
    cvt_f32_f16<<<WN / 1024, 256, 0, stream>>>(up_f32[3], w1u, WN);
    cvt_f32_f16<<<WN / 1024, 256, 0, stream>>>(up_f32[4], w2u, WN);
    cvt_f32_f16<<<WN / 1024, 256, 0, stream>>>(up_f32[5], w3u, WN);
    cvt_f32_f16<<<WN / 1024, 256, 0, stream>>>(dn_f32[3], w1d, WN);
    cvt_f32_f16<<<WN / 1024, 256, 0, stream>>>(dn_f32[4], w2d, WN);
    cvt_f32_f16<<<WN / 1024, 256, 0, stream>>>(dn_f32[5], w3d, WN);
    const int EN = VOCABc * DIMc;
    cvt_f32_f16<<<EN / 1024, 256, 0, stream>>>(emb, embh, EN);

    StackP UP = { up_f32[0], up_f32[1], up_f32[2], up_f32[6], up_f32[7],
                  up_f32[8], up_f32[9], w1u, w2u, w3u };
    StackP DN = { dn_f32[0], dn_f32[1], dn_f32[2], dn_f32[6], dn_f32[7],
                  dn_f32[8], dn_f32[9], w1d, w2d, w3d };

    embed_kernel<<<nEl / 256, 256, 0, stream>>>(idx, emb, pos, t1);
    rmsnorm_kernel<<<MTOK / 8, 256, 0, stream>>>(t1, emb_norm_w, curin, MTOK);
    memgate_kernel<<<(Bc * DIMc) / 256, 256, 0, stream>>>(memory_p, mem_gate_w,
                                                          mem_gate_b, mem);
    shift_add_kernel<<<nEl / 256, 256, 0, stream>>>(curin, token_shift, mem, initial);

    run_stack(initial, UPD, UP, cur, prev, integ, curin, t1, t2h, hidh, stream);
    mix_kernel<<<(Bc * DIMc) / 256, 256, 0, stream>>>(cur);
    sgate_kernel<<<MTOK, 512, 0, stream>>>(cur, initial,
        sg_norm + 0 * DIMc, sg_down_w + 0 * RANKc * DIMc, sg_down_b + 0 * RANKc,
        sg_up_w + 0 * DIMc * RANKc, sg_up_b + 0 * DIMc, g1b);

    run_stack(g1b, DND, DN, cur, prev, integ, curin, t1, t2h, hidh, stream);
    mix_kernel<<<(Bc * DIMc) / 256, 256, 0, stream>>>(cur);
    sgate_kernel<<<MTOK, 512, 0, stream>>>(cur, initial,
        sg_norm + 1 * DIMc, sg_down_w + 1 * RANKc * DIMc, sg_down_b + 1 * RANKc,
        sg_up_w + 1 * DIMc * RANKc, sg_up_b + 1 * DIMc, g2b);

    run_stack(g2b, UPD, UP, cur, prev, integ, curin, t1, t2h, hidh, stream);
    mix_kernel<<<(Bc * DIMc) / 256, 256, 0, stream>>>(cur);
    sgate_kernel<<<MTOK, 512, 0, stream>>>(cur, g2b,
        sg_norm + 2 * DIMc, sg_down_w + 2 * RANKc * DIMc, sg_down_b + 2 * RANKc,
        sg_up_w + 2 * DIMc * RANKc, sg_up_b + 2 * DIMc, ob);

    rmsnorm_f16_kernel<<<MTOK / 8, 256, 0, stream>>>(ob, final_norm_w, t1h, MTOK);
    dim3 gl(MTOK / 32, (VOCABc + 511) / 512);            // (64, 63)
    gemm_logits<<<gl, 256, 0, stream>>>(t1h, embh, (float*)d_out, VOCABc);
}
